// ERSNN_41523743818079
// MI455X (gfx1250) — matrix operand order repaired and checked by reading, not compiled, not run on hardware
//
#include <hip/hip_runtime.h>
#include <stdint.h>

// ---------------------------------------------------------------------------
// SNN (LIF) forward: per timestep 4x [GEMM + fused LIF update], bf16 WMMA.
//   T=25, B=128, D_in=2048, H=4096, C=1000 (padded to 1024)
// Weights converted to bf16 once per call (masked), resident in 192MB L2.
// GEMM: workgroup tile 64M x 128N, wave tile 32M x 32N (2x2 register block),
// A panel staged in LDS per 128-wide K chunk (async-to-LDS when available).
// ---------------------------------------------------------------------------

typedef __attribute__((ext_vector_type(16))) __bf16 v16bf;
typedef __attribute__((ext_vector_type(8)))  __bf16 v8bf;
typedef __attribute__((ext_vector_type(8)))  float  v8f;
typedef __attribute__((ext_vector_type(4)))  int    v4i;

typedef __attribute__((address_space(1))) v4i* glb_v4i_ptr;
typedef __attribute__((address_space(3))) v4i* lds_v4i_ptr;

#if __has_builtin(__builtin_amdgcn_global_load_async_to_lds_b128)
#define HAVE_ASYNC_LDS 1
#else
#define HAVE_ASYNC_LDS 0
#endif

static __device__ __forceinline__ __bf16 f32_to_bf16(float f) {
    unsigned u = __float_as_uint(f);
    unsigned r = (u + 0x7FFFu + ((u >> 16) & 1u)) >> 16;
    return __builtin_bit_cast(__bf16, (unsigned short)r);
}

static __device__ __forceinline__ void wait_async_then_barrier() {
#if HAVE_ASYNC_LDS
#if __has_builtin(__builtin_amdgcn_s_wait_asynccnt)
    __builtin_amdgcn_s_wait_asynccnt(0);
#else
    asm volatile("s_wait_asynccnt 0" ::: "memory");
#endif
#endif
    __syncthreads();
}

// ------------------------------ prep kernels -------------------------------

__global__ void prep_maskmul_bf16(const float* __restrict__ w,
                                  const float* __restrict__ m,
                                  __bf16* __restrict__ out, int n) {
    int i = blockIdx.x * 256 + threadIdx.x;
    if (i < n) out[i] = f32_to_bf16(w[i] * m[i]);
}

__global__ void prep_cvt_bf16(const float* __restrict__ x,
                              __bf16* __restrict__ out, int n) {
    int i = blockIdx.x * 256 + threadIdx.x;
    if (i < n) out[i] = f32_to_bf16(x[i]);
}

// pad w_out [1000,K] -> [1024,K] bf16, zero pad rows
__global__ void prep_wout_bf16(const float* __restrict__ w,
                               __bf16* __restrict__ out,
                               int rows_valid, int K, int total) {
    int i = blockIdx.x * 256 + threadIdx.x;
    if (i < total) {
        int r = i / K;
        out[i] = (r < rows_valid) ? f32_to_bf16(w[i])
                                  : __builtin_bit_cast(__bf16, (unsigned short)0);
    }
}

__global__ void prep_biaspad(const float* __restrict__ b,
                             float* __restrict__ out, int nvalid, int ntotal) {
    int i = blockIdx.x * 256 + threadIdx.x;
    if (i < ntotal) out[i] = (i < nvalid) ? b[i] : 0.0f;
}

__global__ void zero_f32_kernel(float* __restrict__ p, int n) {
    int i = blockIdx.x * 256 + threadIdx.x;
    if (i < n) p[i] = 0.0f;
}

// --------------------------- fused GEMM + LIF ------------------------------
// cur = A[128,K](bf16) @ W[Npad,K](bf16)^T + bias[Npad]
// LIF:  reset = (mem_old > 1);  mem = 0.95*mem_old + cur - reset
//       spk   = (mem > 1) ? 1 : 0
//
// grid.x = Npad/128, grid.y = 128/64 = 2.  256 threads = 8 waves.
// Wave w: mgroup = w>>2 (32 rows), nslice = w&3 (32 cols); 2x2 16x16 tiles.

#define KCHUNK 128

__global__ __launch_bounds__(256)
void snn_gemm_lif(const __bf16* __restrict__ A,
                  const __bf16* __restrict__ W,
                  const float*  __restrict__ bias,
                  float*        __restrict__ mem,
                  __bf16*       __restrict__ spk_out,
                  float*        __restrict__ sum_out,
                  int K, int Npad, int Nout) {
    __shared__ __attribute__((aligned(128))) __bf16 As[64][KCHUNK];

    const int tid    = threadIdx.x;
    const int lane   = tid & 31;
    const int wave   = tid >> 5;
    const int mgroup = wave >> 2;     // 0..1 -> 32-row group
    const int nslice = wave & 3;      // 0..3 -> 32-col slice
    const int mbase  = blockIdx.y * 64;
    const int nbase  = blockIdx.x * 128 + nslice * 32;

    v8f c00 = {}, c01 = {}, c10 = {}, c11 = {};

    // ---- cooperative A->LDS staging: 64x128 bf16 (16KB), 64B per thread ----
    const int eoff = tid * 32;                 // elem offset in 64x128 panel
    const int srow = eoff >> 7;                // 0..63
    const int scol = eoff & 127;               // multiple of 32
    const __bf16* gsrc = A + (size_t)(mbase + srow) * K + scol;
    __bf16* ldst = &As[srow][scol];

    // ---- fragment addressing (contiguous k = 16*(lane>>4) + s, same in A and B) ----
    const int arow  = mgroup * 32 + (lane & 15);     // A frag row, mi=0
    const int akoff = (lane >> 4) * 16;              // A frag K sub-chunk
    const int ncol  = nbase + (lane & 15);           // B frag column, ni=0
    const __bf16* wrow0 = W + (size_t)ncol * K + (lane >> 4) * 16;
    const __bf16* wrow1 = wrow0 + (size_t)16 * K;    // ni=1 (+16 cols)

    for (int kc = 0; kc < K; kc += KCHUNK) {
#if HAVE_ASYNC_LDS
#pragma unroll
        for (int j = 0; j < 4; ++j)
            __builtin_amdgcn_global_load_async_to_lds_b128(
                (glb_v4i_ptr)(uintptr_t)(gsrc + kc + j * 8),
                (lds_v4i_ptr)(unsigned)(uintptr_t)(ldst + j * 8),
                0, 0);
#else
        *(v16bf*)ldst        = *(const v16bf*)(gsrc + kc);
        *(v16bf*)(ldst + 16) = *(const v16bf*)(gsrc + kc + 16);
#endif
        if (kc + KCHUNK < K)
            __builtin_prefetch(wrow0 + (size_t)(kc + KCHUNK), 0, 3);
        wait_async_then_barrier();

#pragma unroll
        for (int kk = 0; kk < KCHUNK; kk += 32) {
            // B fragments (weights, L2-resident), reused across 2 M tiles
            v16bf b0 = *(const v16bf*)(wrow0 + (size_t)(kc + kk));
            v16bf b1 = *(const v16bf*)(wrow1 + (size_t)(kc + kk));
            // A fragments (spikes, LDS), reused across 2 N tiles
            v8bf a0lo = *(const v8bf*)&As[arow][kk + akoff];
            v8bf a0hi = *(const v8bf*)&As[arow][kk + akoff + 8];
            v8bf a1lo = *(const v8bf*)&As[arow + 16][kk + akoff];
            v8bf a1hi = *(const v8bf*)&As[arow + 16][kk + akoff + 8];
            v16bf a0 = __builtin_shufflevector(a0lo, a0hi,
                0, 1, 2, 3, 4, 5, 6, 7, 8, 9, 10, 11, 12, 13, 14, 15);
            v16bf a1 = __builtin_shufflevector(a1lo, a1hi,
                0, 1, 2, 3, 4, 5, 6, 7, 8, 9, 10, 11, 12, 13, 14, 15);
            c00 = __builtin_amdgcn_wmma_f32_16x16x32_bf16(false, a0, false, b0, (short)0, c00, false, false);
            c01 = __builtin_amdgcn_wmma_f32_16x16x32_bf16(false, a0, false, b1, (short)0, c01, false, false);
            c10 = __builtin_amdgcn_wmma_f32_16x16x32_bf16(false, a1, false, b0, (short)0, c10, false, false);
            c11 = __builtin_amdgcn_wmma_f32_16x16x32_bf16(false, a1, false, b1, (short)0, c11, false, false);
        }
        __syncthreads();
    }

    // ---- fused LIF epilogue (C layout: VGPR r -> M = r / r+8, lane -> N) ----
    const __bf16 ONEB  = __builtin_bit_cast(__bf16, (unsigned short)0x3F80);
    const __bf16 ZEROB = __builtin_bit_cast(__bf16, (unsigned short)0x0000);
    const v8f* accs[2][2] = {{&c00, &c01}, {&c10, &c11}};
#pragma unroll
    for (int mi = 0; mi < 2; ++mi) {
#pragma unroll
        for (int ni = 0; ni < 2; ++ni) {
            const v8f& c = *accs[mi][ni];
            const int nc = nbase + ni * 16 + (lane & 15);
            const float bcol = bias[nc];
#pragma unroll
            for (int r = 0; r < 8; ++r) {
                int m = mbase + mgroup * 32 + mi * 16 + r + ((lane >> 4) * 8);
                float cur = c[r] + bcol;
                size_t idx = (size_t)m * Npad + nc;
                float mo = mem[idx];
                float reset = (mo > 1.0f) ? 1.0f : 0.0f;
                float mn = 0.95f * mo + cur - reset;
                mem[idx] = mn;
                float s = (mn > 1.0f) ? 1.0f : 0.0f;
                if (spk_out) spk_out[idx] = (s > 0.5f) ? ONEB : ZEROB;
                if (sum_out && nc < Nout)
                    sum_out[(size_t)m * Nout + nc] += s;
            }
        }
    }
}

// ------------------------------- launcher ----------------------------------

extern "C" void kernel_launch(void* const* d_in, const int* in_sizes, int n_in,
                              void* d_out, int out_size, void* d_ws, size_t ws_size,
                              hipStream_t stream) {
    const float* x_seq = (const float*)d_in[0];
    const float* w1    = (const float*)d_in[1];
    const float* b1    = (const float*)d_in[2];
    const float* m1    = (const float*)d_in[3];
    const float* w2    = (const float*)d_in[4];
    const float* b2    = (const float*)d_in[5];
    const float* m2    = (const float*)d_in[6];
    const float* w3    = (const float*)d_in[7];
    const float* b3    = (const float*)d_in[8];
    const float* m3    = (const float*)d_in[9];
    const float* w_out = (const float*)d_in[10];
    const float* b_out = (const float*)d_in[11];

    constexpr int T = 25, B = 128, DIN = 2048, H = 4096, C = 1000, CP = 1024;

    char* ws = (char*)d_ws;
    auto bump = [&](size_t bytes) -> char* {
        char* p = ws;
        ws += (bytes + 255) & ~(size_t)255;
        return p;
    };
    __bf16* ew1  = (__bf16*)bump((size_t)H * DIN * 2);
    __bf16* ew2  = (__bf16*)bump((size_t)H * H * 2);
    __bf16* ew3  = (__bf16*)bump((size_t)H * H * 2);
    __bf16* ewo  = (__bf16*)bump((size_t)CP * H * 2);
    __bf16* xb   = (__bf16*)bump((size_t)T * B * DIN * 2);
    float*  bo   = (float*)bump((size_t)CP * 4);
    float*  mem1 = (float*)bump((size_t)B * H * 4);   // mem1..memo contiguous
    float*  mem2 = (float*)bump((size_t)B * H * 4);
    float*  mem3 = (float*)bump((size_t)B * H * 4);
    float*  memo = (float*)bump((size_t)B * CP * 4);
    __bf16* spk1 = (__bf16*)bump((size_t)B * H * 2);
    __bf16* spk2 = (__bf16*)bump((size_t)B * H * 2);
    __bf16* spk3 = (__bf16*)bump((size_t)B * H * 2);

    auto nb = [](int n) { return (n + 255) / 256; };

    // ---- prep: masked weights -> bf16, input -> bf16, pad readout ----
    prep_maskmul_bf16<<<nb(H * DIN), 256, 0, stream>>>(w1, m1, ew1, H * DIN);
    prep_maskmul_bf16<<<nb(H * H),   256, 0, stream>>>(w2, m2, ew2, H * H);
    prep_maskmul_bf16<<<nb(H * H),   256, 0, stream>>>(w3, m3, ew3, H * H);
    prep_wout_bf16<<<nb(CP * H), 256, 0, stream>>>(w_out, ewo, C, H, CP * H);
    prep_cvt_bf16<<<nb(T * B * DIN), 256, 0, stream>>>(x_seq, xb, T * B * DIN);
    prep_biaspad<<<nb(CP), 256, 0, stream>>>(b_out, bo, C, CP);

    // ---- zero state (mem1..memo contiguous) and the output accumulator ----
    int n_state = 3 * B * H + B * CP;
    zero_f32_kernel<<<nb(n_state), 256, 0, stream>>>(mem1, n_state);
    zero_f32_kernel<<<nb(B * C), 256, 0, stream>>>((float*)d_out, B * C);

    // ---- time loop: 4 fused GEMM+LIF launches per step ----
    dim3 blk(256);
    dim3 gridH(H / 128, B / 64);    // 32 x 2
    dim3 gridO(CP / 128, B / 64);   //  8 x 2
    for (int t = 0; t < T; ++t) {
        const __bf16* xt = xb + (size_t)t * B * DIN;
        snn_gemm_lif<<<gridH, blk, 0, stream>>>(xt,   ew1, b1, mem1, spk1, nullptr, DIN, H, H);
        snn_gemm_lif<<<gridH, blk, 0, stream>>>(spk1, ew2, b2, mem2, spk2, nullptr, H,   H, H);
        snn_gemm_lif<<<gridH, blk, 0, stream>>>(spk2, ew3, b3, mem3, spk3, nullptr, H,   H, H);
        snn_gemm_lif<<<gridO, blk, 0, stream>>>(spk3, ewo, bo, memo, nullptr, (float*)d_out, H, CP, C);
    }
}
